// HardNegativeMining_6305011991159
// MI455X (gfx1250) — matrix operand order repaired and checked by reading, not compiled, not run on hardware
//
#include <hip/hip_runtime.h>
#include <math.h>

// ---------------------------------------------------------------------------
// Types for CDNA5 WMMA (wave32, v_wmma_f32_16x16x32_f16)
// ---------------------------------------------------------------------------
typedef _Float16 v8h  __attribute__((ext_vector_type(8)));
typedef _Float16 v16h __attribute__((ext_vector_type(16)));
typedef float    v8f  __attribute__((ext_vector_type(8)));

// Problem constants
#define BATCH   4096
#define NNEG    32768
#define MTOT    40960          // 4096 + 4096 + 32768
#define HDIM    1024
#define PDIM    256
#define TOPK    128
#define INV_T   (1.0f / 0.07f)

// Workspace layout (byte offsets). Total < 256 MiB.
#define OFF_Y      ((size_t)0)                  // fp32 Y  [40960][1024]  160 MiB (dies after bn_gelu)
#define OFF_Z      ((size_t)0)                  // fp32 Z  [40960][256]    40 MiB (aliases dead Y)
#define OFF_Z16    ((size_t)64  << 20)          // f16 z   [40960][256]    20 MiB (inside dead Y)
#define OFF_CAND   ((size_t)96  << 20)          // fp32 cand [4096][2048]  32 MiB (inside dead Y)
#define OFF_X16    ((size_t)160 << 20)          // f16 X / Yact [40960][1024] 80 MiB
#define OFF_W1H    ((size_t)240 << 20)          // f16 W1 [1024][1024]      2 MiB
#define OFF_W2H    ((size_t)242 << 20)          // f16 W2 [256][1024]     0.5 MiB
#define OFF_SUM    ((size_t)243 << 20)          // fp32 colsum  [3][1024]  12 KiB
#define OFF_SUM2   (OFF_SUM   + 12288)          // fp32 colsum2 [3][1024]
#define OFF_SCALE  (OFF_SUM2  + 12288)          // fp32 bn scale [3][1024]
#define OFF_SHIFT  (OFF_SCALE + 12288)          // fp32 bn shift [3][1024]
#define OFF_POS    (OFF_SHIFT + 12288)          // fp32 pos [4096]
#define OFF_RLOSS  (OFF_POS   + 16384)          // fp32 per-row loss [4096]

// ---------------------------------------------------------------------------
// WMMA fragment loaders (row-major f16 sources).
// A (16xK tile at (m_base,k0)): lane L holds row m_base+(L&15),
//                               16 contiguous K values starting k0 + (L>>4)*16.
// B (Kx16 tile, B[k][n] = src[n][k]): lane L holds column n_base+(L&15),
//                               16 contiguous K values starting k0 + (L>>4)*16.
// ---------------------------------------------------------------------------
__device__ __forceinline__ v16h load_frag_a(const _Float16* __restrict__ base, int ld,
                                            int m_base, int k0, int lane) {
  int l = lane & 15, hi = lane >> 4;
  const _Float16* p = base + (size_t)(m_base + l) * ld + k0 + hi * 16;
  return *(const v16h*)p;
}

__device__ __forceinline__ v16h load_frag_b(const _Float16* __restrict__ base, int ld,
                                            int n_base, int k0, int lane) {
  int l = lane & 15, hi = lane >> 4;
  const _Float16* p = base + (size_t)(n_base + l) * ld + k0 + hi * 16;
  return *(const v16h*)p;
}

__device__ __forceinline__ v8f wmma_f16(v16h a, v16h b, v8f c) {
  return __builtin_amdgcn_wmma_f32_16x16x32_f16(false, a, false, b, (short)0, c, false, false);
}

// ---------------------------------------------------------------------------
// fp32 -> f16 conversion
// ---------------------------------------------------------------------------
__global__ void cvt_f32_f16(const float* __restrict__ in, _Float16* __restrict__ out, int n) {
  int i = blockIdx.x * blockDim.x + threadIdx.x;
  if (i < n) out[i] = (_Float16)in[i];
}

// ---------------------------------------------------------------------------
// GEMM1: Y[m][n] = sum_k X[m][k]*W1[n][k] + b1[n]; also per-segment column
// sum / sum^2 for BatchNorm (seg 0: h1 rows, 1: h2, 2: h_neg).
// grid (2, 2560), block 256 (8 waves). Wave: 16 rows x 64 cols (4 tiles).
// ---------------------------------------------------------------------------
__global__ void __launch_bounds__(256)
gemm1_kernel(const _Float16* __restrict__ X, const _Float16* __restrict__ W1h,
             const float* __restrict__ b1, float* __restrict__ Y,
             float* __restrict__ colsum, float* __restrict__ colsum2) {
  int lane = threadIdx.x & 31;
  int wave = threadIdx.x >> 5;
  int m_base = blockIdx.y * 16;
  int nb0 = blockIdx.x * 512 + wave * 64;

  v8f acc[4] = {};
  for (int k0 = 0; k0 < HDIM; k0 += 32) {
    v16h a = load_frag_a(X, HDIM, m_base, k0, lane);
    #pragma unroll
    for (int t = 0; t < 4; ++t) {
      v16h b = load_frag_b(W1h, HDIM, nb0 + t * 16, k0, lane);
      acc[t] = wmma_f16(a, b, acc[t]);
    }
  }

  int l = lane & 15, hi = lane >> 4;
  int seg = (m_base < BATCH) ? 0 : (m_base < 2 * BATCH ? 1 : 2);
  #pragma unroll
  for (int t = 0; t < 4; ++t) {
    int n = nb0 + t * 16 + l;
    float bias = b1[n];
    float s = 0.f, s2 = 0.f;
    #pragma unroll
    for (int r = 0; r < 8; ++r) {
      float y = acc[t][r] + bias;
      Y[(size_t)(m_base + r + hi * 8) * HDIM + n] = y;
      s += y; s2 += y * y;
    }
    s  += __shfl_xor(s, 16);
    s2 += __shfl_xor(s2, 16);
    if (hi == 0) {
      atomicAdd(&colsum [seg * HDIM + n], s);
      atomicAdd(&colsum2[seg * HDIM + n], s2);
    }
  }
}

// ---------------------------------------------------------------------------
// BN finalize: biased variance (matches jnp.var), scale/shift per segment.
// ---------------------------------------------------------------------------
__global__ void bn_finalize(const float* __restrict__ colsum, const float* __restrict__ colsum2,
                            const float* __restrict__ gamma, const float* __restrict__ beta,
                            float* __restrict__ scale, float* __restrict__ shift) {
  int idx = blockIdx.x * blockDim.x + threadIdx.x;
  if (idx >= 3 * HDIM) return;
  int seg = idx >> 10, n = idx & (HDIM - 1);
  float cnt = (seg == 2) ? (float)NNEG : (float)BATCH;
  float mu  = colsum[idx] / cnt;
  float var = colsum2[idx] / cnt - mu * mu;
  float sc  = gamma[n] * rsqrtf(var + 1e-5f);
  scale[idx] = sc;
  shift[idx] = beta[n] - mu * sc;
}

// ---------------------------------------------------------------------------
// BN apply + exact-erf GELU + convert to f16 (writes over the dead X16 buf).
// ---------------------------------------------------------------------------
__global__ void bn_gelu(const float* __restrict__ Y, const float* __restrict__ scale,
                        const float* __restrict__ shift, _Float16* __restrict__ Yact) {
  size_t idx = (size_t)blockIdx.x * blockDim.x + threadIdx.x;
  if (idx >= (size_t)MTOT * HDIM) return;
  int row = (int)(idx >> 10);
  int n   = (int)(idx & (HDIM - 1));
  int seg = (row < BATCH) ? 0 : (row < 2 * BATCH ? 1 : 2);
  float y = Y[idx] * scale[seg * HDIM + n] + shift[seg * HDIM + n];
  float g = 0.5f * y * (1.0f + erff(y * 0.70710678118654752440f));
  Yact[idx] = (_Float16)g;
}

// ---------------------------------------------------------------------------
// GEMM2: Z[m][p] = sum_k Yact[m][k]*W2[p][k] + b2[p].  grid 2560, block 256.
// Wave: 16 rows x 32 cols (2 tiles); 8 waves cover all 256 cols.
// ---------------------------------------------------------------------------
__global__ void __launch_bounds__(256)
gemm2_kernel(const _Float16* __restrict__ Yact, const _Float16* __restrict__ W2h,
             const float* __restrict__ b2, float* __restrict__ Z) {
  int lane = threadIdx.x & 31;
  int wave = threadIdx.x >> 5;
  int m_base = blockIdx.x * 16;
  int nb0 = wave * 32;

  v8f acc[2] = {};
  for (int k0 = 0; k0 < HDIM; k0 += 32) {
    v16h a = load_frag_a(Yact, HDIM, m_base, k0, lane);
    #pragma unroll
    for (int t = 0; t < 2; ++t) {
      v16h b = load_frag_b(W2h, HDIM, nb0 + t * 16, k0, lane);
      acc[t] = wmma_f16(a, b, acc[t]);
    }
  }
  int l = lane & 15, hi = lane >> 4;
  #pragma unroll
  for (int t = 0; t < 2; ++t) {
    int n = nb0 + t * 16 + l;
    float bias = b2[n];
    #pragma unroll
    for (int r = 0; r < 8; ++r)
      Z[(size_t)(m_base + r + hi * 8) * PDIM + n] = acc[t][r] + bias;
  }
}

// ---------------------------------------------------------------------------
// L2 normalize rows (256 elems) -> f16.  One wave per row; grid 5120, block 256.
// ---------------------------------------------------------------------------
__global__ void l2norm_kernel(const float* __restrict__ Z, _Float16* __restrict__ z16) {
  int lane = threadIdx.x & 31;
  int row  = blockIdx.x * 8 + (threadIdx.x >> 5);
  const float* zr = Z + (size_t)row * PDIM;
  float v[8], ss = 0.f;
  #pragma unroll
  for (int i = 0; i < 8; ++i) { v[i] = zr[lane + i * 32]; ss += v[i] * v[i]; }
  #pragma unroll
  for (int o = 16; o; o >>= 1) ss += __shfl_xor(ss, o);
  float inv = 1.0f / fmaxf(sqrtf(ss), 1e-12f);
  _Float16* o16 = z16 + (size_t)row * PDIM;
  #pragma unroll
  for (int i = 0; i < 8; ++i) o16[lane + i * 32] = (_Float16)(v[i] * inv);
}

// ---------------------------------------------------------------------------
// pos[b] = z1[b] . z2[b]  (f16 rows, fp32 accumulate). One wave per row.
// ---------------------------------------------------------------------------
__global__ void pos_kernel(const _Float16* __restrict__ z16, float* __restrict__ pos) {
  int lane = threadIdx.x & 31;
  int b    = blockIdx.x * 8 + (threadIdx.x >> 5);
  const _Float16* a = z16 + (size_t)b * PDIM;
  const _Float16* c = z16 + (size_t)(BATCH + b) * PDIM;
  float d = 0.f;
  for (int i = lane; i < PDIM; i += 32) d += (float)a[i] * (float)c[i];
  #pragma unroll
  for (int o = 16; o; o >>= 1) d += __shfl_xor(d, o);
  if (lane == 0) pos[b] = d;
}

// ---------------------------------------------------------------------------
// In-LDS block-wide bitonic sort, descending, n = power of two, 256 threads.
// ---------------------------------------------------------------------------
__device__ __forceinline__ void bitonic_sort_desc(float* a, int n) {
  for (int k = 2; k <= n; k <<= 1) {
    for (int j = k >> 1; j > 0; j >>= 1) {
      for (int i = threadIdx.x; i < n; i += 256) {
        int ixj = i ^ j;
        if (ixj > i) {
          float x = a[i], y = a[ixj];
          bool upper = (i & k) != 0;
          bool sw = upper ? (x > y) : (x < y);   // descending
          if (sw) { a[i] = y; a[ixj] = x; }
        }
      }
      __syncthreads();
    }
  }
}

// ---------------------------------------------------------------------------
// Stage 1: sim tile 16 rows x 2048 negatives via WMMA into LDS (128 KB),
// per-row bitonic sort, emit per-block top-128 candidates.
// grid (16 col-blocks, 256 row-blocks), block 256, dyn LDS = 16*2048*4.
// ---------------------------------------------------------------------------
__global__ void __launch_bounds__(256)
sim_topk_stage1(const _Float16* __restrict__ z16, float* __restrict__ cand) {
  extern __shared__ float sim[];                 // [16][2048]
  int lane = threadIdx.x & 31;
  int wave = threadIdx.x >> 5;
  int m_base = blockIdx.y * 16;                  // z1 rows
  int cbase  = blockIdx.x * 2048;                // negative block
  const _Float16* z1 = z16;                                  // rows [0,4096)
  const _Float16* zn = z16 + (size_t)(2 * BATCH) * PDIM;     // z_neg rows

  // Preload all A fragments for this row block (K = 256 -> 8 frags).
  v16h afrag[8];
  #pragma unroll
  for (int kk = 0; kk < 8; ++kk) afrag[kk] = load_frag_a(z1, PDIM, m_base, kk * 32, lane);

  int l = lane & 15, hi = lane >> 4;
  for (int t = 0; t < 16; ++t) {                 // wave covers 256 cols = 16 tiles
    int n_local = wave * 256 + t * 16;
    v8f acc = {};
    #pragma unroll
    for (int kk = 0; kk < 8; ++kk) {
      v16h b = load_frag_b(zn, PDIM, cbase + n_local, kk * 32, lane);
      acc = wmma_f16(afrag[kk], b, acc);
    }
    #pragma unroll
    for (int r = 0; r < 8; ++r)
      sim[(r + hi * 8) * 2048 + n_local + l] = acc[r];
  }
  __syncthreads();

  for (int row = 0; row < 16; ++row) {
    float* a = sim + row * 2048;
    bitonic_sort_desc(a, 2048);
    if (threadIdx.x < TOPK)
      cand[(size_t)(m_base + row) * 2048 + blockIdx.x * TOPK + threadIdx.x] = a[threadIdx.x];
    __syncthreads();
  }
}

// ---------------------------------------------------------------------------
// Stage 2: per row, sort the 2048 candidates, take top-128, logsumexp with
// the positive logit -> per-row loss.  grid 4096, dyn LDS = 2048*4.
// ---------------------------------------------------------------------------
__global__ void __launch_bounds__(256)
topk_stage2(const float* __restrict__ cand, const float* __restrict__ pos,
            float* __restrict__ rowloss) {
  extern __shared__ float a[];                   // 2048
  __shared__ float red[8];
  int b = blockIdx.x;
  for (int i = threadIdx.x; i < 2048; i += 256) a[i] = cand[(size_t)b * 2048 + i];
  __syncthreads();
  bitonic_sort_desc(a, 2048);

  float p    = pos[b] * INV_T;
  float vmax = fmaxf(p, a[0] * INV_T);
  float part = 0.f;
  if (threadIdx.x < TOPK) part = expf(a[threadIdx.x] * INV_T - vmax);
  #pragma unroll
  for (int o = 16; o; o >>= 1) part += __shfl_xor(part, o);
  if ((threadIdx.x & 31) == 0) red[threadIdx.x >> 5] = part;
  __syncthreads();
  if (threadIdx.x == 0) {
    float total = expf(p - vmax);
    #pragma unroll
    for (int w = 0; w < 8; ++w) total += red[w];
    rowloss[b] = vmax + logf(total) - p;         // -log_softmax[0]
  }
}

// ---------------------------------------------------------------------------
// Final: mean of 4096 per-row losses.
// ---------------------------------------------------------------------------
__global__ void final_mean(const float* __restrict__ rowloss, float* __restrict__ out) {
  __shared__ float red[8];
  float s = 0.f;
  for (int i = threadIdx.x; i < BATCH; i += 256) s += rowloss[i];
  #pragma unroll
  for (int o = 16; o; o >>= 1) s += __shfl_xor(s, o);
  if ((threadIdx.x & 31) == 0) red[threadIdx.x >> 5] = s;
  __syncthreads();
  if (threadIdx.x == 0) {
    float t = 0.f;
    #pragma unroll
    for (int w = 0; w < 8; ++w) t += red[w];
    out[0] = t / (float)BATCH;
  }
}

// ---------------------------------------------------------------------------
// Launch
// ---------------------------------------------------------------------------
extern "C" void kernel_launch(void* const* d_in, const int* in_sizes, int n_in,
                              void* d_out, int out_size, void* d_ws, size_t ws_size,
                              hipStream_t stream) {
  (void)in_sizes; (void)n_in; (void)out_size; (void)ws_size;
  const float* h1    = (const float*)d_in[0];
  const float* h2    = (const float*)d_in[1];
  const float* hneg  = (const float*)d_in[2];
  const float* W1    = (const float*)d_in[3];
  const float* b1    = (const float*)d_in[4];
  const float* gamma = (const float*)d_in[5];
  const float* beta  = (const float*)d_in[6];
  const float* W2    = (const float*)d_in[7];
  const float* b2    = (const float*)d_in[8];

  char* ws = (char*)d_ws;
  float*    Y       = (float*)   (ws + OFF_Y);
  float*    Z       = (float*)   (ws + OFF_Z);
  _Float16* z16     = (_Float16*)(ws + OFF_Z16);
  float*    cand    = (float*)   (ws + OFF_CAND);
  _Float16* X16     = (_Float16*)(ws + OFF_X16);   // also Yact16 after bn_gelu
  _Float16* W1h     = (_Float16*)(ws + OFF_W1H);
  _Float16* W2h     = (_Float16*)(ws + OFF_W2H);
  float*    colsum  = (float*)   (ws + OFF_SUM);
  float*    colsum2 = (float*)   (ws + OFF_SUM2);
  float*    scale   = (float*)   (ws + OFF_SCALE);
  float*    shift   = (float*)   (ws + OFF_SHIFT);
  float*    pos     = (float*)   (ws + OFF_POS);
  float*    rloss   = (float*)   (ws + OFF_RLOSS);

  // Zero BN accumulators every call (deterministic; graph-capturable).
  hipMemsetAsync(ws + OFF_SUM, 0, 2 * 12288, stream);

  // fp32 -> f16 conversions.
  cvt_f32_f16<<<(BATCH * HDIM + 255) / 256, 256, 0, stream>>>(h1,   X16,                       BATCH * HDIM);
  cvt_f32_f16<<<(BATCH * HDIM + 255) / 256, 256, 0, stream>>>(h2,   X16 + (size_t)BATCH * HDIM, BATCH * HDIM);
  cvt_f32_f16<<<(NNEG  * HDIM + 255) / 256, 256, 0, stream>>>(hneg, X16 + (size_t)2 * BATCH * HDIM, NNEG * HDIM);
  cvt_f32_f16<<<(HDIM  * HDIM + 255) / 256, 256, 0, stream>>>(W1, W1h, HDIM * HDIM);
  cvt_f32_f16<<<(PDIM  * HDIM + 255) / 256, 256, 0, stream>>>(W2, W2h, PDIM * HDIM);

  // Projection head.
  gemm1_kernel<<<dim3(2, MTOT / 16), 256, 0, stream>>>(X16, W1h, b1, Y, colsum, colsum2);
  bn_finalize<<<(3 * HDIM + 255) / 256, 256, 0, stream>>>(colsum, colsum2, gamma, beta, scale, shift);
  bn_gelu<<<(int)(((size_t)MTOT * HDIM + 255) / 256), 256, 0, stream>>>(Y, scale, shift, X16);
  gemm2_kernel<<<MTOT / 16, 256, 0, stream>>>(X16, W2h, b2, Z);
  l2norm_kernel<<<MTOT / 8, 256, 0, stream>>>(Z, z16);

  // Loss pieces.
  pos_kernel<<<BATCH / 8, 256, 0, stream>>>(z16, pos);
  sim_topk_stage1<<<dim3(NNEG / 2048, BATCH / 16), 256, 16 * 2048 * sizeof(float), stream>>>(z16, cand);
  topk_stage2<<<BATCH, 256, 2048 * sizeof(float), stream>>>(cand, pos, rloss);
  final_mean<<<1, 256, 0, stream>>>(rloss, (float*)d_out);
}
